// NodeEncoder_29188597743891
// MI455X (gfx1250) — matrix operand order repaired and checked by reading, not compiled, not run on hardware
//
#include <hip/hip_runtime.h>
#include <hip/hip_fp16.h>

typedef __attribute__((ext_vector_type(16))) _Float16 v16h;
typedef __attribute__((ext_vector_type(8)))  _Float16 v8h;
typedef __attribute__((ext_vector_type(4)))  _Float16 v4h;
typedef __attribute__((ext_vector_type(8)))  float    v8f;

#define DD 128   // embed dim
#define HH 4     // heads
#define HDIM 32  // per-head dim

// ----------------------------------------------------------------------------
// WMMA GEMM: out[M,N] = epi(A[M,K] @ W[K,N] + bias[N])  (opt ReLU, opt +resid)
// K compile-time, K % 32 == 0, N % 64 == 0.
// Block = 256 threads = 8 wave32 waves; tile BM=64 x BN=64; wave = 16(M) x 32(N)
// (one A fragment reused across two B fragments -> 2 WMMAs per K-step).
// Branch-free staging: row-clamped float4 loads, fp32->fp16 packed LDS stores.
// ----------------------------------------------------------------------------
template<int K>
__global__ __launch_bounds__(256)
void gemm_bias_kernel(const float* __restrict__ A, const float* __restrict__ W,
                      const float* __restrict__ bias, const float* __restrict__ resid,
                      float* __restrict__ out, int M, int N, int relu)
{
    constexpr int BM = 64, BN = 64, BK = 32;
    constexpr int LDA = 40;                       // padded k-stride (halves), 80B rows
    __shared__ __align__(16) _Float16 As[BM * LDA];   // [m][k]
    __shared__ __align__(16) _Float16 Bs[BN * LDA];   // transposed [n][k]

    const int tid      = threadIdx.x;
    const int wave     = tid >> 5;
    const int lane     = tid & 31;
    const int wm       = wave & 3;        // M offset 16*wm
    const int wn       = wave >> 2;       // N offset 32*wn
    const int m0       = blockIdx.x * BM;
    const int n0       = blockIdx.y * BN;
    const int laneHalf = lane >> 4;       // 0 | 1
    const int lane15   = lane & 15;

    v8f acc0 = {}, acc1 = {};

    for (int k0 = 0; k0 < K; k0 += BK) {
        // ---- stage A tile: 64x32 fp32 -> fp16, 2 x float4 per thread ----
        #pragma unroll
        for (int i = 0; i < 2; ++i) {
            int idx4 = tid + i * 256;             // 0..511 float4 slots
            int r = idx4 >> 3;                    // 0..63
            int c = (idx4 & 7) * 4;               // 0..28
            int gm = m0 + r; gm = gm < M ? gm : M - 1;   // clamp: masked at store
            float4 av = *(const float4*)&A[(long)gm * K + k0 + c];
            v4h hv; hv[0] = (_Float16)av.x; hv[1] = (_Float16)av.y;
                    hv[2] = (_Float16)av.z; hv[3] = (_Float16)av.w;
            *(v4h*)&As[r * LDA + c] = hv;
        }
        // ---- stage B tile transposed: Bs[n][k] = W[k0+k][n0+n], 2 x float4 ----
        #pragma unroll
        for (int i = 0; i < 2; ++i) {
            int idx4 = tid + i * 256;             // 32(k) x 16 float4(n)
            int kr = idx4 >> 4;                   // 0..31
            int c  = (idx4 & 15) * 4;             // 0..60
            float4 wv = *(const float4*)&W[(long)(k0 + kr) * N + n0 + c];
            Bs[(c + 0) * LDA + kr] = (_Float16)wv.x;
            Bs[(c + 1) * LDA + kr] = (_Float16)wv.y;
            Bs[(c + 2) * LDA + kr] = (_Float16)wv.z;
            Bs[(c + 3) * LDA + kr] = (_Float16)wv.w;
        }
        __syncthreads();

        // ---- A fragment (16x32 f16) ----
        const int am = wm * 16 + lane15;
        const int kb = laneHalf * 16;
        v8h alo = *(const v8h*)&As[am * LDA + kb];
        v8h ahi = *(const v8h*)&As[am * LDA + kb + 8];
        v16h afrag;
        #pragma unroll
        for (int i = 0; i < 8; ++i) { afrag[i] = alo[i]; afrag[i + 8] = ahi[i]; }

        // ---- two B fragments (32x16 f16 each) ----
        const int kb2 = laneHalf * 16;
        const int bnA = wn * 32 + lane15;
        const int bnB = bnA + 16;
        v8h b0lo = *(const v8h*)&Bs[bnA * LDA + kb2];
        v8h b0hi = *(const v8h*)&Bs[bnA * LDA + kb2 + 8];
        v8h b1lo = *(const v8h*)&Bs[bnB * LDA + kb2];
        v8h b1hi = *(const v8h*)&Bs[bnB * LDA + kb2 + 8];
        v16h bf0, bf1;
        #pragma unroll
        for (int i = 0; i < 8; ++i) {
            bf0[i] = b0lo[i]; bf0[i + 8] = b0hi[i];
            bf1[i] = b1lo[i]; bf1[i + 8] = b1hi[i];
        }

        acc0 = __builtin_amdgcn_wmma_f32_16x16x32_f16(false, afrag, false, bf0,
                                                      (short)0, acc0, false, false);
        acc1 = __builtin_amdgcn_wmma_f32_16x16x32_f16(false, afrag, false, bf1,
                                                      (short)0, acc1, false, false);
        __syncthreads();
    }

    // ---- epilogue: VGPR r -> M = r + 8*(lane>=16), N = lane&15 ----
    const int gn0   = n0 + wn * 32 + lane15;
    const int gn1   = gn0 + 16;
    const int mbase = m0 + wm * 16 + laneHalf * 8;
    const float bv0 = bias[gn0];
    const float bv1 = bias[gn1];
    #pragma unroll
    for (int r = 0; r < 8; ++r) {
        int gm = mbase + r;
        if (gm < M) {
            float o0 = acc0[r] + bv0;
            float o1 = acc1[r] + bv1;
            if (relu) { o0 = fmaxf(o0, 0.0f); o1 = fmaxf(o1, 0.0f); }
            if (resid) {
                o0 += resid[(long)gm * N + gn0];
                o1 += resid[(long)gm * N + gn1];
            }
            out[(long)gm * N + gn0] = o0;
            out[(long)gm * N + gn1] = o1;
        }
    }
}

// ----------------------------------------------------------------------------
// Layer 1 of prep MLP: out[M,64] = relu(x[M,5] @ W1[5,64] + b1). K=5 -> VALU.
// W1/b1 cached in LDS; 64 threads share one x row (L1 broadcast).
// ----------------------------------------------------------------------------
__global__ __launch_bounds__(256)
void prep1_kernel(const float* __restrict__ x, const float* __restrict__ W1,
                  const float* __restrict__ b1, float* __restrict__ out, int M)
{
    __shared__ float sW[5 * 64];
    __shared__ float sB[64];
    if (threadIdx.x < 5 * 64) sW[threadIdx.x] = W1[threadIdx.x];
    if (threadIdx.x < 64)     sB[threadIdx.x] = b1[threadIdx.x];
    __syncthreads();

    int t = blockIdx.x * blockDim.x + threadIdx.x;
    int m = t >> 6, n = t & 63;
    if (m >= M) return;
    const float* xr = x + (long)m * 5;
    float acc = sB[n];
    #pragma unroll
    for (int k = 0; k < 5; ++k) acc += xr[k] * sW[k * 64 + n];
    out[(long)m * 64 + n] = fmaxf(acc, 0.0f);
}

// ----------------------------------------------------------------------------
// Per-(edge,head) attention logit -> exp(score/sqrt(32)); accumulate denom.
// ----------------------------------------------------------------------------
__global__ __launch_bounds__(256)
void edge_score_kernel(const float* __restrict__ q, const float* __restrict__ k,
                       const int* __restrict__ src, const int* __restrict__ dst,
                       float* __restrict__ ex, float* __restrict__ denom, int E)
{
    int t = blockIdx.x * blockDim.x + threadIdx.x;
    if (t >= E * HH) return;
    int eid = t >> 2, h = t & 3;
    int s = src[eid], d = dst[eid];
    const float4* qp = (const float4*)(q + (long)d * DD + h * HDIM);
    const float4* kp = (const float4*)(k + (long)s * DD + h * HDIM);
    float acc = 0.0f;
    #pragma unroll
    for (int i = 0; i < HDIM / 4; ++i) {
        float4 a = qp[i], b = kp[i];
        acc += a.x * b.x + a.y * b.y + a.z * b.z + a.w * b.w;
    }
    float e = expf(acc * 0.17677669529663687f);   // 1/sqrt(32)
    ex[t] = e;
    atomicAdd(&denom[(long)d * HH + h], e);
}

// ----------------------------------------------------------------------------
// Scatter-aggregate: agg[dst] += alpha * v[src].  One thread per 4 floats.
// ----------------------------------------------------------------------------
__global__ __launch_bounds__(256)
void edge_agg_kernel(const float* __restrict__ v, const float* __restrict__ ex,
                     const float* __restrict__ denom,
                     const int* __restrict__ src, const int* __restrict__ dst,
                     float* __restrict__ agg, int E)
{
    long t = (long)blockIdx.x * blockDim.x + threadIdx.x;
    if (t >= (long)E * 32) return;
    int eid = (int)(t >> 5);
    int d0  = ((int)t & 31) * 4;          // 4 contiguous dims within one head
    int h   = d0 >> 5;
    int s = src[eid], d = dst[eid];
    float alpha = ex[(long)eid * HH + h] / denom[(long)d * HH + h];
    float4 vv = *(const float4*)(v + (long)s * DD + d0);
    float* ap = agg + (long)d * DD + d0;
    atomicAdd(&ap[0], alpha * vv.x);
    atomicAdd(&ap[1], alpha * vv.y);
    atomicAdd(&ap[2], alpha * vv.z);
    atomicAdd(&ap[3], alpha * vv.w);
}

__global__ __launch_bounds__(256)
void zero_kernel(float* __restrict__ a, long n)
{
    long i = (long)blockIdx.x * blockDim.x + threadIdx.x;
    long stride = (long)gridDim.x * blockDim.x;
    for (; i < n; i += stride) a[i] = 0.0f;
}

__global__ __launch_bounds__(256)
void add_inplace_kernel(float* __restrict__ a, const float* __restrict__ b, long n)
{
    long i = (long)blockIdx.x * blockDim.x + threadIdx.x;
    if (i < n) a[i] += b[i];
}

static inline void launch_gemm(const float* A, const float* W, const float* bias,
                               const float* resid, float* out,
                               int M, int K, int N, int relu, hipStream_t stream)
{
    dim3 grid((M + 63) / 64, N / 64);
    if (K == 64)
        gemm_bias_kernel<64><<<grid, 256, 0, stream>>>(A, W, bias, resid, out, M, N, relu);
    else
        gemm_bias_kernel<128><<<grid, 256, 0, stream>>>(A, W, bias, resid, out, M, N, relu);
}

extern "C" void kernel_launch(void* const* d_in, const int* in_sizes, int n_in,
                              void* d_out, int out_size, void* d_ws, size_t ws_size,
                              hipStream_t stream)
{
    const float* x       = (const float*)d_in[0];
    const int*   eix     = (const int*)  d_in[1];   // [2,E]: row0=src, row1=dst
    const float* prep_W1 = (const float*)d_in[2];
    const float* prep_b1 = (const float*)d_in[3];
    const float* prep_W2 = (const float*)d_in[4];
    const float* prep_b2 = (const float*)d_in[5];
    const float* prep_W3 = (const float*)d_in[6];
    const float* prep_b3 = (const float*)d_in[7];
    const float* upd_W1  = (const float*)d_in[8];
    const float* upd_b1  = (const float*)d_in[9];
    const float* upd_W2  = (const float*)d_in[10];
    const float* upd_b2  = (const float*)d_in[11];
    const float* upd_W3  = (const float*)d_in[12];
    const float* upd_b3  = (const float*)d_in[13];
    const float* q_W     = (const float*)d_in[14];
    const float* q_b     = (const float*)d_in[15];
    const float* k_W     = (const float*)d_in[16];
    const float* k_b     = (const float*)d_in[17];
    const float* v_W     = (const float*)d_in[18];
    const float* v_b     = (const float*)d_in[19];
    const float* skip_W  = (const float*)d_in[20];
    const float* skip_b  = (const float*)d_in[21];

    const int N = in_sizes[0] / 5;        // nodes
    const int E = in_sizes[1] / 2;        // edges
    const int H1 = 64, H2 = 128;
    const int* src = eix;
    const int* dst = eix + E;

    // Workspace layout (floats)
    float* ws    = (float*)d_ws;
    long   off   = 0;
    float* bufA  = ws + off; off += (long)N * H1;   // h1 / u1
    float* bufB  = ws + off; off += (long)N * DD;   // h2 / u2
    float* hinit = ws + off; off += (long)N * DD;
    float* q     = ws + off; off += (long)N * DD;
    float* k     = ws + off; off += (long)N * DD;
    float* v     = ws + off; off += (long)N * DD;
    float* skip  = ws + off; off += (long)N * DD;
    float* ex    = ws + off; off += (long)E * HH;
    float* denom = ws + off; off += (long)N * HH;
    float* agg   = ws + off; off += (long)N * DD;   // later h = agg + skip (in place)

    // Zero accumulation buffers
    zero_kernel<<<2048, 256, 0, stream>>>(denom, (long)N * HH);
    zero_kernel<<<4096, 256, 0, stream>>>(agg, (long)N * DD);

    // prep MLP: x -> 64 -> 128 -> D
    prep1_kernel<<<(N * 64 + 255) / 256, 256, 0, stream>>>(x, prep_W1, prep_b1, bufA, N);
    launch_gemm(bufA, prep_W2, prep_b2, nullptr, bufB,  N, H1, H2, 1, stream);
    launch_gemm(bufB, prep_W3, prep_b3, nullptr, hinit, N, H2, DD, 0, stream);

    // q/k/v/skip projections
    launch_gemm(hinit, q_W,    q_b,    nullptr, q,    N, DD, DD, 0, stream);
    launch_gemm(hinit, k_W,    k_b,    nullptr, k,    N, DD, DD, 0, stream);
    launch_gemm(hinit, v_W,    v_b,    nullptr, v,    N, DD, DD, 0, stream);
    launch_gemm(hinit, skip_W, skip_b, nullptr, skip, N, DD, DD, 0, stream);

    // edge attention: exp(logits) + segment denom, then scatter-aggregate
    {
        int total = E * HH;
        edge_score_kernel<<<(total + 255) / 256, 256, 0, stream>>>(q, k, src, dst, ex, denom, E);
        long tot2 = (long)E * 32;
        edge_agg_kernel<<<(int)((tot2 + 255) / 256), 256, 0, stream>>>(v, ex, denom, src, dst, agg, E);
    }

    // h = agg + skip   (in place -> agg holds h)
    {
        long n = (long)N * DD;
        add_inplace_kernel<<<(int)((n + 255) / 256), 256, 0, stream>>>(agg, skip, n);
    }

    // update MLP with residual: out = h + mlp(h)
    launch_gemm(agg,  upd_W1, upd_b1, nullptr, bufA, N, DD, H1, 1, stream);
    launch_gemm(bufA, upd_W2, upd_b2, nullptr, bufB, N, H1, H2, 1, stream);
    launch_gemm(bufB, upd_W3, upd_b3, agg, (float*)d_out, N, H2, DD, 0, stream);
}
